// SuperSwinTransformerBlock_71975061947033
// MI455X (gfx1250) — matrix operand order repaired and checked by reading, not compiled, not run on hardware
//
#include <hip/hip_runtime.h>

// ---------------------------------------------------------------------------
// Swin block, MI455X (gfx1250, wave32).  All GEMMs via v_wmma_f32_16x16x32_bf16.
// Window-tile staging via TDM tensor_load_to_lds (fallback: b128 copy).
// ---------------------------------------------------------------------------

typedef __bf16 bf16;
typedef __attribute__((ext_vector_type(16))) __bf16 bf16x16;
typedef __attribute__((ext_vector_type(8)))  __bf16 bf16x8;
typedef __attribute__((ext_vector_type(8)))  float  f32x8;
typedef __attribute__((ext_vector_type(4)))  unsigned int u32x4;
typedef __attribute__((ext_vector_type(8)))  int i32x8;
typedef __attribute__((ext_vector_type(4)))  int i32x4;

#define BATCH 64
#define HH    56
#define CCH   128
#define NHD   4
#define HD    32
#define WSZ   7
#define SHF   3
#define NTOK  49
#define NPAD  64
#define HID   512
#define NWIN  4096          // BATCH * 64 windows
#define SCALE 0.1767766953f // 32^-0.5

__device__ __forceinline__ f32x8 wmma_bf16(bf16x16 a, bf16x16 b, f32x8 c) {
  return __builtin_amdgcn_wmma_f32_16x16x32_bf16(false, a, false, b, (short)0, c,
                                                 false, false);
}

// A fragment: 16x32 bf16, row-major source with leading dim lda (elements).
__device__ __forceinline__ bf16x16 load_a_frag(const bf16* base, int lda, int lane) {
  int m  = lane & 15;
  int kh = (lane >> 4) << 4;             // 0 or 16
  const bf16* p = base + m * lda + kh;
  union { bf16x16 v; bf16x8 h[2]; } u;
  u.h[0] = *(const bf16x8*)(p);          // K = kh .. kh+7
  u.h[1] = *(const bf16x8*)(p + 8);      // K = kh+8 .. kh+15
  return u.v;
}

// B fragment: 32x16 bf16; source is B^T row-major (N x K), leading dim ldb.
__device__ __forceinline__ bf16x16 load_b_frag(const bf16* baseT, int ldb, int lane) {
  int n  = lane & 15;
  int k0 = (lane >> 4) << 4;             // 0 or 16
  return *(const bf16x16*)(baseT + n * ldb + k0);
}

__device__ __forceinline__ int tok_cnt(int r, int c, int wh, int ww) {
  int hs = wh * 7 + r, ws = ww * 7 + c;
  int rh = (hs < 49) ? 0 : ((hs < 53) ? 1 : 2);
  int rw = (ws < 49) ? 0 : ((ws < 53) ? 1 : 2);
  return rh * 3 + rw;
}

// ---------------------------------------------------------------------------
// K0: f32 -> bf16 weight conversion
// ---------------------------------------------------------------------------
__global__ void k_cvt(const float* __restrict__ src, bf16* __restrict__ dst, int n) {
  int i = blockIdx.x * 256 + threadIdx.x;
  if (i < n) dst[i] = (bf16)src[i];
}

// ---------------------------------------------------------------------------
// K1: LN1 + cyclic shift + window partition -> xwin bf16 (NWIN, 64, 128)
// ---------------------------------------------------------------------------
__global__ __launch_bounds__(256) void k_prep(const float* __restrict__ x,
                                              const float* __restrict__ gamma,
                                              const float* __restrict__ beta,
                                              bf16* __restrict__ xwin) {
  int row  = blockIdx.x * 8 + (threadIdx.x >> 5);
  int lane = threadIdx.x & 31;
  int wid  = row >> 6;
  int n    = row & 63;
  bf16* dst = xwin + (size_t)row * CCH;
  if (n >= NTOK) {                       // zero padding rows
    *(float2*)(dst + lane * 4) = make_float2(0.f, 0.f);
    return;
  }
  int b  = wid >> 6;
  int wi = wid & 63;
  int wh = wi >> 3, ww = wi & 7;
  int r = n / 7, c = n - r * 7;
  int h0 = (wh * 7 + r + SHF) % HH;
  int w0 = (ww * 7 + c + SHF) % HH;
  const float4* xp = (const float4*)(x + ((size_t)b * (HH * HH) + h0 * HH + w0) * CCH);
  float4 v = xp[lane];
  float s = v.x + v.y + v.z + v.w;
#pragma unroll
  for (int off = 16; off > 0; off >>= 1) s += __shfl_xor(s, off, 32);
  float mu = s * (1.f / 128.f);
  float d0 = v.x - mu, d1 = v.y - mu, d2 = v.z - mu, d3 = v.w - mu;
  float vs = d0 * d0 + d1 * d1 + d2 * d2 + d3 * d3;
#pragma unroll
  for (int off = 16; off > 0; off >>= 1) vs += __shfl_xor(vs, off, 32);
  float rstd = rsqrtf(vs * (1.f / 128.f) + 1e-5f);
  float4 gg = ((const float4*)gamma)[lane];
  float4 bb = ((const float4*)beta)[lane];
  dst[lane * 4 + 0] = (bf16)(d0 * rstd * gg.x + bb.x);
  dst[lane * 4 + 1] = (bf16)(d1 * rstd * gg.y + bb.y);
  dst[lane * 4 + 2] = (bf16)(d2 * rstd * gg.z + bb.z);
  dst[lane * 4 + 3] = (bf16)(d3 * rstd * gg.w + bb.w);
}

// ---------------------------------------------------------------------------
// K2: per-window fused attention. 1 block (8 waves) per window.
// ---------------------------------------------------------------------------
__global__ __launch_bounds__(256) void k_attn(const bf16* __restrict__ xwin,
                                              const bf16* __restrict__ wq,
                                              const float* __restrict__ qkv_b,
                                              const float* __restrict__ rpb,
                                              const bf16* __restrict__ wp,
                                              const float* __restrict__ proj_b,
                                              const float* __restrict__ xin,
                                              float* __restrict__ out) {
  __shared__ __align__(32) bf16 smem[32768];   // 64 KB
  bf16* sX  = smem;           // [64][128], later reused as sO
  bf16* sQ  = smem + 8192;    // [4][64][32]  (scaled q)
  bf16* sK  = smem + 16384;   // [4][64][32]
  bf16* sVT = smem + 24576;   // [4][32][64]  (v transposed)
  bf16* sP  = smem + 8192;    // aliases sQ+sK after barrier: 8 waves x [32][64]
  bf16* sO  = sX;             // aliases sX after barrier

  int wv   = threadIdx.x >> 5;
  int lane = threadIdx.x & 31;
  int lg   = lane >> 4;
  int wid  = blockIdx.x;
  int bimg = wid >> 6;
  int wi   = wid & 63;
  int wh   = wi >> 3, ww = wi & 7;

  __builtin_prefetch(wq, 0, 1);
  __builtin_prefetch(wp, 0, 1);

  // ---- stage window tile (64x128 bf16 = 16 KB) into LDS via TDM ----
#if __has_builtin(__builtin_amdgcn_tensor_load_to_lds) && \
    __has_builtin(__builtin_amdgcn_s_wait_tensorcnt)
  if (wv == 0) {
    unsigned long long ga = (unsigned long long)(const void*)(xwin + (size_t)wid * NPAD * CCH);
    unsigned int lds = (unsigned int)(unsigned long long)(const void*)sX;
    // D# group0: count=1 | lds_addr | global_addr[56:0] | type=2
    u32x4 g0;
    g0.x = 1u;
    g0.y = lds;
    g0.z = (unsigned int)(ga & 0xffffffffu);
    g0.w = (unsigned int)((ga >> 32) & 0x01ffffffu) | 0x80000000u;
    // D# group1: data_size=2B; 1 row x 8192 elems; tile_dim0=8192, tile_dim1=1
    i32x8 g1;
    g1[0] = 0x00010000;          // data_size=1 (2 bytes), mask=0
    g1[1] = 0x20000000;          // tensor_dim0 (=8192) low bits at [63:48]
    g1[2] = 0x00010000;          // tensor_dim0 hi=0, tensor_dim1=1 at [95:80]
    g1[3] = 0x20000000;          // tile_dim0 = 8192 at [127:112]
    g1[4] = 1;                   // tile_dim1 = 1 at [143:128]
    g1[5] = 8192;                // tensor_dim0_stride at [191:160]
    g1[6] = 0;
    g1[7] = 0;
    i32x4 gz4 = {0, 0, 0, 0};
    i32x8 gz8 = {0, 0, 0, 0, 0, 0, 0, 0};
    __builtin_amdgcn_tensor_load_to_lds(g0, g1, gz4, gz4, gz8, 0);
    __builtin_amdgcn_s_wait_tensorcnt((short)0);
  }
#else
  {
    const int4* src = (const int4*)(xwin + (size_t)wid * NPAD * CCH);
    int4* dst = (int4*)sX;
#pragma unroll
    for (int i = 0; i < 4; ++i) dst[threadIdx.x + 256 * i] = src[threadIdx.x + 256 * i];
  }
#endif
  __syncthreads();

  // ---- QKV GEMM: (64x128) x (128x384) ----
#pragma unroll
  for (int j = 0; j < 3; ++j) {
    int nt = wv + 8 * j;                        // 0..23 (wave-uniform)
    int s3 = nt >> 3;                           // 0=q 1=k 2=v  (uniform)
    int h  = (nt >> 1) & 3;                     // head          (uniform)
    int dlo = ((nt & 1) << 4) + (lane & 15);    // d within head (per-lane)
    float bia = qkv_b[nt * 16 + (lane & 15)];
#pragma unroll
    for (int mt = 0; mt < 4; ++mt) {
      f32x8 acc = {};
#pragma unroll
      for (int kb = 0; kb < 4; ++kb) {
        bf16x16 a = load_a_frag(sX + mt * 16 * CCH + kb * 32, CCH, lane);
        bf16x16 b = load_b_frag(wq + (nt * 16) * CCH + kb * 32, CCH, lane);
        acc = wmma_bf16(a, b, acc);
      }
      if (s3 == 0) {
#pragma unroll
        for (int r = 0; r < 8; ++r) {
          int m = mt * 16 + r + 8 * lg;
          sQ[(h * 64 + m) * 32 + dlo] = (bf16)((acc[r] + bia) * SCALE);
        }
      } else if (s3 == 1) {
#pragma unroll
        for (int r = 0; r < 8; ++r) {
          int m = mt * 16 + r + 8 * lg;
          sK[(h * 64 + m) * 32 + dlo] = (bf16)(acc[r] + bia);
        }
      } else {
        bf16x8 pk;
#pragma unroll
        for (int r = 0; r < 8; ++r) pk[r] = (bf16)(acc[r] + bia);
        *(bf16x8*)(sVT + (h * 32 + dlo) * 64 + mt * 16 + 8 * lg) = pk;  // 8 tokens
      }
    }
  }
  __syncthreads();

  // ---- S = Q K^T : wave wv -> head h, query half mh ----
  int h  = wv >> 1;
  int mh = wv & 1;
  f32x8 sacc[2][4];
#pragma unroll
  for (int mi = 0; mi < 2; ++mi) {
    bf16x16 a = load_a_frag(sQ + (h * 64 + mh * 32 + mi * 16) * 32, 32, lane);
#pragma unroll
    for (int ni = 0; ni < 4; ++ni) {
      bf16x16 b = load_b_frag(sK + (h * 64 + ni * 16) * 32, 32, lane);
      f32x8 z = {};
      sacc[mi][ni] = wmma_bf16(a, b, z);
    }
  }
  __syncthreads();   // all sQ/sK reads done; sP may now overwrite them

  // ---- rel-pos bias + shift mask + padding mask ----
  int rk4[4], ck4[4], cnk4[4];
  bool kv4[4];
#pragma unroll
  for (int ni = 0; ni < 4; ++ni) {
    int nk = ni * 16 + (lane & 15);
    kv4[ni] = nk < NTOK;
    int rk = nk / 7, ck = nk - rk * 7;
    rk4[ni] = rk; ck4[ni] = ck;
    cnk4[ni] = kv4[ni] ? tok_cnt(rk, ck, wh, ww) : -1;
  }
#pragma unroll
  for (int mi = 0; mi < 2; ++mi) {
#pragma unroll
    for (int r = 0; r < 8; ++r) {
      int mq = mh * 32 + mi * 16 + r + 8 * lg;
      bool qv = mq < NTOK;
      int rq = mq / 7, cq = mq - rq * 7;
      int cntq = qv ? tok_cnt(rq, cq, wh, ww) : -1;
#pragma unroll
      for (int ni = 0; ni < 4; ++ni) {
        float sv = sacc[mi][ni][r];
        if (qv && kv4[ni]) {
          int idx = (rq - rk4[ni] + 6) * 13 + (cq - ck4[ni] + 6);
          sv += rpb[idx * NHD + h];
          if (cntq != cnk4[ni]) sv -= 100.f;
        } else {
          sv = -1e30f;
        }
        sacc[mi][ni][r] = sv;
      }
    }
  }

  // ---- softmax (row reductions across 16 lanes of each half-wave) ----
  bf16* sPw = sP + wv * (32 * 64);
#pragma unroll
  for (int mi = 0; mi < 2; ++mi) {
#pragma unroll
    for (int r = 0; r < 8; ++r) {
      float v0 = sacc[mi][0][r], v1 = sacc[mi][1][r];
      float v2 = sacc[mi][2][r], v3 = sacc[mi][3][r];
      float mx = fmaxf(fmaxf(v0, v1), fmaxf(v2, v3));
#pragma unroll
      for (int off = 1; off < 16; off <<= 1) mx = fmaxf(mx, __shfl_xor(mx, off, 32));
      float e0 = __expf(v0 - mx), e1 = __expf(v1 - mx);
      float e2 = __expf(v2 - mx), e3 = __expf(v3 - mx);
      float sum = e0 + e1 + e2 + e3;
#pragma unroll
      for (int off = 1; off < 16; off <<= 1) sum += __shfl_xor(sum, off, 32);
      float inv = 1.f / sum;
      int rowl = mi * 16 + r + 8 * lg;
      int col = lane & 15;
      sPw[rowl * 64 +  0 + col] = (bf16)(e0 * inv);
      sPw[rowl * 64 + 16 + col] = (bf16)(e1 * inv);
      sPw[rowl * 64 + 32 + col] = (bf16)(e2 * inv);
      sPw[rowl * 64 + 48 + col] = (bf16)(e3 * inv);
    }
  }

  // ---- O = P V ----
#pragma unroll
  for (int mi = 0; mi < 2; ++mi) {
#pragma unroll
    for (int ni = 0; ni < 2; ++ni) {
      f32x8 acc = {};
#pragma unroll
      for (int kb = 0; kb < 2; ++kb) {
        bf16x16 a = load_a_frag(sPw + (mi * 16) * 64 + kb * 32, 64, lane);
        bf16x16 b = load_b_frag(sVT + (h * 32 + ni * 16) * 64 + kb * 32, 64, lane);
        acc = wmma_bf16(a, b, acc);
      }
      int d = ni * 16 + (lane & 15);
#pragma unroll
      for (int r = 0; r < 8; ++r) {
        int m = mh * 32 + mi * 16 + r + 8 * lg;
        sO[m * CCH + h * 32 + d] = (bf16)acc[r];
      }
    }
  }
  __syncthreads();

  // ---- proj GEMM (64x128)x(128x128) + residual, reverse shift scatter ----
  {
    int nt = wv;
    int cf = nt * 16 + (lane & 15);
    float pb = proj_b[cf];
#pragma unroll
    for (int mt = 0; mt < 4; ++mt) {
      f32x8 acc = {};
#pragma unroll
      for (int kb = 0; kb < 4; ++kb) {
        bf16x16 a = load_a_frag(sO + mt * 16 * CCH + kb * 32, CCH, lane);
        bf16x16 b = load_b_frag(wp + (nt * 16) * CCH + kb * 32, CCH, lane);
        acc = wmma_bf16(a, b, acc);
      }
#pragma unroll
      for (int r = 0; r < 8; ++r) {
        int tok = mt * 16 + r + 8 * lg;
        if (tok < NTOK) {
          int rr = tok / 7, cc = tok - rr * 7;
          int hf = (wh * 7 + rr + SHF) % HH;
          int wf = (ww * 7 + cc + SHF) % HH;
          size_t base = ((size_t)bimg * (HH * HH) + hf * HH + wf) * CCH + cf;
          out[base] = xin[base] + acc[r] + pb;
        }
      }
    }
  }
}

// ---------------------------------------------------------------------------
// K3: LN2 + MLP (128 -> 512 -> 128, exact GELU) + residual, in place on d_out.
// ---------------------------------------------------------------------------
__global__ __launch_bounds__(256) void k_mlp(float* __restrict__ xio,
                                             const float* __restrict__ g2,
                                             const float* __restrict__ b2,
                                             const bf16* __restrict__ w1,
                                             const float* __restrict__ b1f,
                                             const bf16* __restrict__ w2,
                                             const float* __restrict__ b2f) {
  __shared__ __align__(32) bf16 sA[32 * CCH];   // 8 KB  (LN2 output)
  __shared__ __align__(32) bf16 sH[32 * HID];   // 32 KB (GELU(fc1) output)

  int wv   = threadIdx.x >> 5;
  int lane = threadIdx.x & 31;
  int lg   = lane >> 4;
  size_t tok0 = (size_t)blockIdx.x * 32;

  __builtin_prefetch(w1, 0, 1);
  __builtin_prefetch(w2, 0, 1);

  // ---- LN2 into sA ----
#pragma unroll
  for (int i = 0; i < 4; ++i) {
    int lt = wv + 8 * i;
    const float4* xp = (const float4*)(xio + (tok0 + lt) * CCH);
    float4 v = xp[lane];
    float s = v.x + v.y + v.z + v.w;
#pragma unroll
    for (int off = 16; off > 0; off >>= 1) s += __shfl_xor(s, off, 32);
    float mu = s * (1.f / 128.f);
    float d0 = v.x - mu, d1 = v.y - mu, d2 = v.z - mu, d3 = v.w - mu;
    float vs = d0 * d0 + d1 * d1 + d2 * d2 + d3 * d3;
#pragma unroll
    for (int off = 16; off > 0; off >>= 1) vs += __shfl_xor(vs, off, 32);
    float rstd = rsqrtf(vs * (1.f / 128.f) + 1e-5f);
    float4 gg = ((const float4*)g2)[lane];
    float4 bb = ((const float4*)b2)[lane];
    bf16* dst = sA + lt * CCH + lane * 4;
    dst[0] = (bf16)(d0 * rstd * gg.x + bb.x);
    dst[1] = (bf16)(d1 * rstd * gg.y + bb.y);
    dst[2] = (bf16)(d2 * rstd * gg.z + bb.z);
    dst[3] = (bf16)(d3 * rstd * gg.w + bb.w);
  }
  __syncthreads();

  // ---- fc1 GEMM (32x128)x(128x512) + exact GELU ----
#pragma unroll
  for (int j = 0; j < 4; ++j) {
    int nt = wv * 4 + j;                        // 0..31
    int f = nt * 16 + (lane & 15);
    float bb = b1f[f];
#pragma unroll
    for (int mt = 0; mt < 2; ++mt) {
      f32x8 acc = {};
#pragma unroll
      for (int kb = 0; kb < 4; ++kb) {
        bf16x16 a = load_a_frag(sA + mt * 16 * CCH + kb * 32, CCH, lane);
        bf16x16 b = load_b_frag(w1 + (nt * 16) * CCH + kb * 32, CCH, lane);
        acc = wmma_bf16(a, b, acc);
      }
#pragma unroll
      for (int r = 0; r < 8; ++r) {
        int m = mt * 16 + r + 8 * lg;
        float v = acc[r] + bb;
        float gel = 0.5f * v * (1.f + erff(v * 0.70710678f));
        sH[m * HID + f] = (bf16)gel;
      }
    }
  }
  __syncthreads();

  // ---- fc2 GEMM (32x512)x(512x128) + residual (in-place) ----
  {
    int nt = wv;                                // 0..7
    int cf = nt * 16 + (lane & 15);
    float bb = b2f[cf];
#pragma unroll
    for (int mt = 0; mt < 2; ++mt) {
      f32x8 acc = {};
#pragma unroll
      for (int kb = 0; kb < 16; ++kb) {
        bf16x16 a = load_a_frag(sH + mt * 16 * HID + kb * 32, HID, lane);
        bf16x16 b = load_b_frag(w2 + (nt * 16) * HID + kb * 32, HID, lane);
        acc = wmma_bf16(a, b, acc);
      }
#pragma unroll
      for (int r = 0; r < 8; ++r) {
        int m = mt * 16 + r + 8 * lg;
        size_t idx = (tok0 + m) * CCH + cf;
        xio[idx] = xio[idx] + acc[r] + bb;      // residual: x + mlp(ln2(x))
      }
    }
  }
}

// ---------------------------------------------------------------------------
extern "C" void kernel_launch(void* const* d_in, const int* in_sizes, int n_in,
                              void* d_out, int out_size, void* d_ws, size_t ws_size,
                              hipStream_t stream) {
  const float* x      = (const float*)d_in[0];
  const float* ln1_g  = (const float*)d_in[1];
  const float* ln1_b  = (const float*)d_in[2];
  const float* qkv_w  = (const float*)d_in[3];
  const float* qkv_b  = (const float*)d_in[4];
  const float* rpb    = (const float*)d_in[5];
  const float* proj_w = (const float*)d_in[6];
  const float* proj_b = (const float*)d_in[7];
  const float* ln2_g  = (const float*)d_in[8];
  const float* ln2_b  = (const float*)d_in[9];
  const float* fc1_w  = (const float*)d_in[10];
  const float* fc1_b  = (const float*)d_in[11];
  const float* fc2_w  = (const float*)d_in[12];
  const float* fc2_b  = (const float*)d_in[13];
  float* out = (float*)d_out;

  char* ws = (char*)d_ws;
  bf16* wq   = (bf16*)(ws + 0);        // 384*128  bf16 =  98304 B
  bf16* wp   = (bf16*)(ws + 98304);    // 128*128  bf16 =  32768 B
  bf16* w1   = (bf16*)(ws + 131072);   // 512*128  bf16 = 131072 B
  bf16* w2   = (bf16*)(ws + 262144);   // 128*512  bf16 = 131072 B
  bf16* xwin = (bf16*)(ws + 393216);   // 4096*64*128 bf16 = 64 MB

  k_cvt<<<(49152 + 255) / 256, 256, 0, stream>>>(qkv_w, wq, 49152);
  k_cvt<<<(16384 + 255) / 256, 256, 0, stream>>>(proj_w, wp, 16384);
  k_cvt<<<(65536 + 255) / 256, 256, 0, stream>>>(fc1_w, w1, 65536);
  k_cvt<<<(65536 + 255) / 256, 256, 0, stream>>>(fc2_w, w2, 65536);

  k_prep<<<NWIN * NPAD / 8, 256, 0, stream>>>(x, ln1_g, ln1_b, xwin);
  k_attn<<<NWIN, 256, 0, stream>>>(xwin, wq, qkv_b, rpb, wp, proj_b, x, out);
  k_mlp<<<(BATCH * HH * HH) / 32, 256, 0, stream>>>(out, ln2_g, ln2_b, w1, fc1_b,
                                                    w2, fc2_b);
}
